// StickBreakingAttention_16810501997098
// MI455X (gfx1250) — matrix operand order repaired and checked by reading, not compiled, not run on hardware
//
#include <hip/hip_runtime.h>

// ---------- CDNA5 WMMA types ----------
typedef __bf16  v16bf __attribute__((ext_vector_type(16)));
typedef float   v8f   __attribute__((ext_vector_type(8)));

struct Frag {
    union { v16bf v; uint4 q[2]; };
};

__device__ __forceinline__ v8f wmma_bf16(const Frag& a, const Frag& b, v8f c) {
    return __builtin_amdgcn_wmma_f32_16x16x32_bf16(false, a.v, false, b.v,
                                                   (short)0, c, false, false);
}

__device__ __forceinline__ unsigned short f2bf(float f) {
    unsigned u = __float_as_uint(f);
    u += 0x7FFFu + ((u >> 16) & 1u);   // round-to-nearest-even
    return (unsigned short)(u >> 16);
}

// ---------- CDNA5 async global->LDS copy (ASYNCcnt-tracked, no VGPR staging) ----------
__device__ __forceinline__ void async_cp_b128(const unsigned short* lds_dst,
                                              const unsigned short* g_src) {
    unsigned lds = (unsigned)(size_t)lds_dst;   // low 32 bits = LDS byte offset
    asm volatile("global_load_async_to_lds_b128 %0, %1, off"
                 :: "v"(lds), "v"(g_src) : "memory");
}
__device__ __forceinline__ void wait_async0() {
    asm volatile("s_wait_asynccnt 0x0" ::: "memory");
}

#define EDIM 2048
#define NH   32
#define HD   64
#define TT   1024

// ---------- Kernel 0a: fp32 -> bf16 convert (hidden states) ----------
__global__ __launch_bounds__(256) void convert_bf16_kernel(const float* __restrict__ src,
                                                           unsigned short* __restrict__ dst,
                                                           int n4) {
    int i = blockIdx.x * 256 + threadIdx.x;
    if (i >= n4) return;
    float4 v = ((const float4*)src)[i];
    unsigned lo = (unsigned)f2bf(v.x) | ((unsigned)f2bf(v.y) << 16);
    unsigned hi = (unsigned)f2bf(v.z) | ((unsigned)f2bf(v.w) << 16);
    ((uint2*)dst)[i] = make_uint2(lo, hi);
}

// ---------- Kernel 0b: transpose+convert weights: Wt[n][k] = bf16(W[k][n]) ----------
__global__ __launch_bounds__(256) void transpose_w_kernel(const float* __restrict__ W0,
                                                          const float* __restrict__ W1,
                                                          const float* __restrict__ W2,
                                                          const float* __restrict__ W3,
                                                          unsigned short* __restrict__ WT) {
    const float* W = (blockIdx.z == 0) ? W0 : (blockIdx.z == 1) ? W1
                   : (blockIdx.z == 2) ? W2 : W3;
    unsigned short* Out = WT + (size_t)blockIdx.z * ((size_t)EDIM * EDIM);
    __shared__ unsigned short tile[64][65];
    const int tid = threadIdx.x;
    const int k0 = blockIdx.y * 64, n0 = blockIdx.x * 64;
#pragma unroll
    for (int i = 0; i < 4; ++i) {
        int fidx = tid + i * 256;              // float4 index within 64x64 tile
        int kl = fidx >> 4;
        int nl = (fidx & 15) << 2;
        float4 v = *(const float4*)(W + (size_t)(k0 + kl) * EDIM + n0 + nl);
        tile[nl + 0][kl] = f2bf(v.x);
        tile[nl + 1][kl] = f2bf(v.y);
        tile[nl + 2][kl] = f2bf(v.z);
        tile[nl + 3][kl] = f2bf(v.w);
    }
    __syncthreads();
#pragma unroll
    for (int i = 0; i < 4; ++i) {
        int oidx = tid + i * 256;              // ushort4 index
        int nl = oidx >> 4;
        int kl = (oidx & 15) << 2;
        unsigned lo = (unsigned)tile[nl][kl]     | ((unsigned)tile[nl][kl + 1] << 16);
        unsigned hi = (unsigned)tile[nl][kl + 2] | ((unsigned)tile[nl][kl + 3] << 16);
        *(uint2*)(Out + (size_t)(n0 + nl) * EDIM + k0 + kl) = make_uint2(lo, hi);
    }
}

// ---------- GEMM: C[2048x2048] = A(bf16,[m][k]) * Bt(bf16,[n][k])^T ----------
// LDS double-buffered with GLOBAL_LOAD_ASYNC_TO_LDS_B128 staging: the next
// K-slab streams into the alternate buffer (no VGPR staging, no ds_stores,
// no spills) while the current slab's 16 WMMAs run; one s_wait_asynccnt 0
// + one barrier per K-step.
// mode 0: write Qbf [b,h,t,d]  mode 1: Kbf [b,h,t,d]
// mode 2: Vt  [b,h,d,t]        mode 3: fp32 Out [row][col]
__global__ __launch_bounds__(256) void gemm_bf16_kernel(const unsigned short* __restrict__ A,
                                                        const unsigned short* __restrict__ Bt,
                                                        void* __restrict__ Out, int mode) {
    __shared__ __align__(16) unsigned short As[2][128 * 64];
    __shared__ __align__(16) unsigned short Bs[2][128 * 64];

    const int tid  = threadIdx.x;
    const int wave = tid >> 5;
    const int lane = tid & 31;
    const int ln16 = lane & 15;
    const int hi   = lane >> 4;
    const int rw   = wave & 3;        // row group: rows rw*32 .. +31
    const int cw   = wave >> 2;       // col group: cols cw*64 .. +63
    const int rowBase = blockIdx.y * 128;
    const int colBase = blockIdx.x * 128;

    const int sRow = tid >> 1;            // 0..127
    const int sOff = (tid & 1) * 32;      // bf16 offset in 64-wide k slab

    const unsigned short* gA = A  + (size_t)(rowBase + sRow) * EDIM + sOff;
    const unsigned short* gB = Bt + (size_t)(colBase + sRow) * EDIM + sOff;

    // prologue: async-stage slab 0 into buffer 0
#pragma unroll
    for (int x = 0; x < 4; ++x) {
        async_cp_b128(As[0] + sRow * 64 + sOff + x * 8, gA + x * 8);
        async_cp_b128(Bs[0] + sRow * 64 + sOff + x * 8, gB + x * 8);
    }
    wait_async0();
    __syncthreads();

    v8f acc[2][4];
#pragma unroll
    for (int ri = 0; ri < 2; ++ri)
#pragma unroll
        for (int ci = 0; ci < 4; ++ci) acc[ri][ci] = (v8f)0.f;

    int p = 0;
    for (int i = 0; i < EDIM / 64; ++i) {
        // stream next slab into the alternate buffer (overlaps with WMMAs)
        if (i < EDIM / 64 - 1) {
            const unsigned short* nA = gA + (size_t)(i + 1) * 64;
            const unsigned short* nB = gB + (size_t)(i + 1) * 64;
#pragma unroll
            for (int x = 0; x < 4; ++x) {
                async_cp_b128(As[p ^ 1] + sRow * 64 + sOff + x * 8, nA + x * 8);
                async_cp_b128(Bs[p ^ 1] + sRow * 64 + sOff + x * 8, nB + x * 8);
            }
        }

        const unsigned short* as = As[p];
        const unsigned short* bs = Bs[p];
#pragma unroll
        for (int kk = 0; kk < 64; kk += 32) {
            Frag a[2], b[4];
#pragma unroll
            for (int ri = 0; ri < 2; ++ri) {
                const int fRow = rw * 32 + ri * 16 + ln16;
                a[ri].q[0] = *(const uint4*)(as + fRow * 64 + kk + hi * 8);
                a[ri].q[1] = *(const uint4*)(as + fRow * 64 + kk + hi * 8 + 16);
            }
#pragma unroll
            for (int ci = 0; ci < 4; ++ci) {
                const int bRow = cw * 64 + ci * 16 + ln16;
                b[ci].q[0] = *(const uint4*)(bs + bRow * 64 + kk + hi * 8);
                b[ci].q[1] = *(const uint4*)(bs + bRow * 64 + kk + hi * 8 + 16);
            }
#pragma unroll
            for (int ri = 0; ri < 2; ++ri)
#pragma unroll
                for (int ci = 0; ci < 4; ++ci)
                    acc[ri][ci] = wmma_bf16(a[ri], b[ci], acc[ri][ci]);
        }

        wait_async0();       // drain this wave's async copies
        __syncthreads();     // all waves drained -> alternate buffer valid
        p ^= 1;
    }

#pragma unroll
    for (int ri = 0; ri < 2; ++ri) {
#pragma unroll
        for (int ci = 0; ci < 4; ++ci) {
#pragma unroll
            for (int r = 0; r < 8; ++r) {
                const int row = rowBase + rw * 32 + ri * 16 + hi * 8 + r;
                const int col = colBase + cw * 64 + ci * 16 + ln16;
                const float v = acc[ri][ci][r];
                if (mode == 3) {
                    ((float*)Out)[(size_t)row * EDIM + col] = v;
                } else {
                    const int bb = row >> 10, t = row & 1023;
                    const int h  = col >> 6,  d = col & 63;
                    const unsigned short bv = f2bf(v);
                    if (mode == 2)  // V transposed: [bh][d][t]
                        ((unsigned short*)Out)[(((size_t)(bb * NH + h) * HD + d) << 10) + t] = bv;
                    else            // Q/K: [bh][t][d]
                        ((unsigned short*)Out)[(((size_t)(bb * NH + h) * TT + t) * HD) + d] = bv;
                }
            }
        }
    }
}

// ---------- Stick-breaking attention: one wave per (16-query tile, head, batch) ----------
__global__ __launch_bounds__(32) void sb_attn_kernel(const unsigned short* __restrict__ Qb,
                                                     const unsigned short* __restrict__ Kb,
                                                     const unsigned short* __restrict__ Vt,
                                                     unsigned short* __restrict__ Ob) {
    __shared__ __align__(16) unsigned short attn_s[16 * 32];  // 1 KB, wave-private

    const int lane = threadIdx.x & 31;
    const int ln16 = lane & 15;
    const int hi   = lane >> 4;
    const int it   = blockIdx.x;        // query tile 0..63
    const int h    = blockIdx.y;
    const int b    = blockIdx.z;
    const int bh   = b * NH + h;

    const unsigned short* Qp = Qb + (size_t)bh * TT * HD;
    const unsigned short* Kp = Kb + (size_t)bh * TT * HD;
    const unsigned short* Vp = Vt + (size_t)bh * HD * TT;

    // Q A-fragments (16 queries x 64 d, split in two K=32 fragments)
    Frag qf[2];
    {
        const int row = it * 16 + ln16;
#pragma unroll
        for (int kk = 0; kk < 2; ++kk) {
            const int kb = kk * 32 + hi * 8;
            qf[kk].q[0] = *(const uint4*)(Qp + (size_t)row * HD + kb);
            qf[kk].q[1] = *(const uint4*)(Qp + (size_t)row * HD + kb + 16);
        }
    }

    v8f oacc[4];
#pragma unroll
    for (int n = 0; n < 4; ++n) oacc[n] = (v8f)0.f;
    float S[8];
#pragma unroll
    for (int r = 0; r < 8; ++r) S[r] = 0.f;

    // descending key chunks of 32 (stick-breaking suffix scan order)
    for (int c = it >> 1; c >= 0; --c) {
        const int cb = c * 32;
        for (int sub = 1; sub >= 0; --sub) {
            const int kb = cb + sub * 16;
            // K B-fragments: lane = key column, d in table order
            Frag kf[2];
            const int krow = kb + ln16;
#pragma unroll
            for (int kk = 0; kk < 2; ++kk) {
                const int ko = kk * 32 + hi * 8;
                kf[kk].q[0] = *(const uint4*)(Kp + (size_t)krow * HD + ko);
                kf[kk].q[1] = *(const uint4*)(Kp + (size_t)krow * HD + ko + 16);
            }
            v8f z = (v8f)0.f;
            z = wmma_bf16(qf[0], kf[0], z);
            z = wmma_bf16(qf[1], kf[1], z);

            const int j = kb + ln16;              // this lane's key index
#pragma unroll
            for (int r = 0; r < 8; ++r) {
                const int  qrow = it * 16 + hi * 8 + r;
                const float zz  = 0.125f * z[r];  // 1/sqrt(64)
                const bool caus = (j < qrow);
                const float sp  = log1pf(expf(-fabsf(zz)));
                const float lb  = fminf(zz, 0.f) - sp;   // log sigmoid(z)
                const float lo  = -fmaxf(zz, 0.f) - sp;  // log sigmoid(-z)
                const float mo  = caus ? lo : 0.f;
                // inclusive suffix scan over the 16-lane segment (descending key)
                float incl = mo;
#pragma unroll
                for (int off = 1; off < 16; off <<= 1) {
                    float t = __shfl_down(incl, off, 16);
                    incl += ((ln16 + off) < 16) ? t : 0.f;
                }
                const float tot   = __shfl(incl, 0, 16);       // whole-subtile sum
                const float sexcl = incl - mo + S[r];          // exclusive + carried suffix
                const float a     = caus ? expf(lb + sexcl) : 0.f;
                S[r] += tot;
                attn_s[(hi * 8 + r) * 32 + sub * 16 + ln16] = f2bf(a);
            }
        }
        __syncthreads();
        // attn A-fragment [16 q x 32 keys]
        Frag af;
        af.q[0] = *(const uint4*)(attn_s + ln16 * 32 + hi * 8);
        af.q[1] = *(const uint4*)(attn_s + ln16 * 32 + hi * 8 + 16);
#pragma unroll
        for (int n = 0; n < 4; ++n) {
            Frag vf;   // V B-fragment: lane = d column, keys in table order (Vt layout)
            const int d = n * 16 + ln16;
            vf.q[0] = *(const uint4*)(Vp + (size_t)d * TT + cb + hi * 8);
            vf.q[1] = *(const uint4*)(Vp + (size_t)d * TT + cb + hi * 8 + 16);
            oacc[n] = wmma_bf16(af, vf, oacc[n]);
        }
        __syncthreads();
    }

    // write attention output as bf16 [token][E] (col = h*64 + d)
#pragma unroll
    for (int n = 0; n < 4; ++n) {
#pragma unroll
        for (int r = 0; r < 8; ++r) {
            const int tok = b * TT + it * 16 + hi * 8 + r;
            const int col = h * HD + n * 16 + ln16;
            Ob[(size_t)tok * EDIM + col] = f2bf(oacc[n][r]);
        }
    }
}

// ---------- Host launcher ----------
extern "C" void kernel_launch(void* const* d_in, const int* in_sizes, int n_in,
                              void* d_out, int out_size, void* d_ws, size_t ws_size,
                              hipStream_t stream) {
    (void)in_sizes; (void)n_in; (void)out_size; (void)ws_size;
    const float* hs = (const float*)d_in[0];
    const float* Wq = (const float*)d_in[1];
    const float* Wk = (const float*)d_in[2];
    const float* Wv = (const float*)d_in[3];
    const float* Wo = (const float*)d_in[4];

    const size_t MAT = (size_t)EDIM * EDIM;        // 4M elements
    unsigned short* Hbf = (unsigned short*)d_ws;                       // 8 MB
    unsigned short* WT  = Hbf + MAT;                                   // 4 x 8 MB
    unsigned short* Qbf = WT  + 4 * MAT;                               // 8 MB
    unsigned short* Kbf = Qbf + MAT;                                   // 8 MB
    unsigned short* Vt  = Kbf + MAT;                                   // 8 MB
    unsigned short* Obf = Vt  + MAT;                                   // 8 MB  (72 MB total)

    // 1) hidden -> bf16
    const int n4 = (int)(MAT / 4);
    convert_bf16_kernel<<<dim3((n4 + 255) / 256), dim3(256), 0, stream>>>(hs, Hbf, n4);
    // 2) weights -> transposed bf16
    transpose_w_kernel<<<dim3(32, 32, 4), dim3(256), 0, stream>>>(Wq, Wk, Wv, Wo, WT);
    // 3) Q/K/V projections (bf16 WMMA)
    gemm_bf16_kernel<<<dim3(16, 16), dim3(256), 0, stream>>>(Hbf, WT + 0 * MAT, Qbf, 0);
    gemm_bf16_kernel<<<dim3(16, 16), dim3(256), 0, stream>>>(Hbf, WT + 1 * MAT, Kbf, 1);
    gemm_bf16_kernel<<<dim3(16, 16), dim3(256), 0, stream>>>(Hbf, WT + 2 * MAT, Vt, 2);
    // 4) stick-breaking attention
    sb_attn_kernel<<<dim3(64, NH, 2), dim3(32), 0, stream>>>(Qbf, Kbf, Vt, Obf);
    // 5) output projection -> fp32 d_out
    gemm_bf16_kernel<<<dim3(16, 16), dim3(256), 0, stream>>>(Obf, WT + 3 * MAT, d_out, 3);
}
